// BetaGNN_12790412607971
// MI455X (gfx1250) — compile-verified, matrix operand order checked by reading, not run on hardware
//
#include <hip/hip_runtime.h>

typedef __attribute__((ext_vector_type(16))) __bf16 v16bf;
typedef __attribute__((ext_vector_type(8)))  float  v8f;

#define HIDDIM 256

// ---------------------------------------------------------------------------
// Fragment loader (CDNA5 wave32 WMMA 16-bit layout, A 16x32 / B 32x16, both
// K-contiguous in memory):
//   lane L: row = L%16 ; K-half = (L>>4)*8 ; VGPRs v0..3 -> K = kh+0..7,
//   v4..7 -> K = 16+kh+0..7  (two contiguous 16-byte runs -> global_load_b128)
// ---------------------------------------------------------------------------
__device__ __forceinline__ v16bf frag_bf16(const __bf16* __restrict__ base,
                                           int row0, int k0, int lane) {
  int r  = row0 + (lane & 15);
  int kh = (lane >> 4) << 3;
  const __bf16* p = base + (size_t)r * HIDDIM + k0 + kh;
  v16bf f;
#pragma unroll
  for (int j = 0; j < 8; ++j) f[j]     = p[j];
#pragma unroll
  for (int j = 0; j < 8; ++j) f[8 + j] = p[16 + j];
  return f;
}

// ---------------------------------------------------------------------------
// fp32 -> bf16 matrix conversion (8 elements / thread)
// ---------------------------------------------------------------------------
__global__ void cvt_f32_bf16(const float* __restrict__ in, __bf16* __restrict__ out,
                             long long n) {
  long long i = ((long long)blockIdx.x * blockDim.x + threadIdx.x) * 8;
  if (i >= n) return;
#pragma unroll
  for (int j = 0; j < 8; ++j) out[i + j] = (__bf16)in[i + j];
}

__global__ void cvt_weights(const float* __restrict__ w0, const float* __restrict__ w1,
                            const float* __restrict__ w2,
                            __bf16* __restrict__ o0, __bf16* __restrict__ o1,
                            __bf16* __restrict__ o2) {
  int i = blockIdx.x * blockDim.x + threadIdx.x;   // grid covers 65536
  o0[i] = (__bf16)w0[i];
  o1[i] = (__bf16)w1[i];
  o2[i] = (__bf16)w2[i];
}

// ---------------------------------------------------------------------------
// Hbf[M,256] = bf16(relu(Abf @ W^T + bias))
// grid = (mblocks, 2); blockIdx.y selects a 128-column half.
// Each wave: 16 rows x 128 cols = 8 WMMA accumulators, K streamed once.
// ---------------------------------------------------------------------------
__global__ void __launch_bounds__(256)
gemm1_relu_bias(const __bf16* __restrict__ A, const __bf16* __restrict__ W,
                const float* __restrict__ bias, __bf16* __restrict__ C, int M) {
  int lane  = threadIdx.x & 31;
  int mtile = blockIdx.x * 8 + (threadIdx.x >> 5);
  int m0    = mtile << 4;
  if (m0 >= M) return;
  int n0 = blockIdx.y << 7;                       // 0 or 128

  v8f acc[8] = {v8f{}, v8f{}, v8f{}, v8f{}, v8f{}, v8f{}, v8f{}, v8f{}};
#pragma unroll
  for (int k0 = 0; k0 < HIDDIM; k0 += 32) {
    v16bf a = frag_bf16(A, m0, k0, lane);
#pragma unroll
    for (int t = 0; t < 8; ++t) {
      v16bf b = frag_bf16(W, n0 + t * 16, k0, lane);
      acc[t] = __builtin_amdgcn_wmma_f32_16x16x32_bf16(
          false, a, false, b, (short)0, acc[t], false, false);
    }
  }
#pragma unroll
  for (int t = 0; t < 8; ++t) {
    int n = n0 + t * 16 + (lane & 15);
    float bv = bias[n];
#pragma unroll
    for (int v = 0; v < 8; ++v) {
      int m = m0 + ((lane >> 4) << 3) + v;
      float x = acc[t][v] + bv;
      C[(size_t)m * HIDDIM + n] = (__bf16)(x > 0.f ? x : 0.f);
    }
  }
}

// ---------------------------------------------------------------------------
// H2[M,256] = relu(AHbf @ W1^T + A2Hbf @ W2^T)   (fp32 output for the GEMV)
// ---------------------------------------------------------------------------
__global__ void __launch_bounds__(256)
gemm2_relu(const __bf16* __restrict__ A1, const __bf16* __restrict__ A2,
           const __bf16* __restrict__ W1, const __bf16* __restrict__ W2,
           float* __restrict__ C, int M) {
  int lane  = threadIdx.x & 31;
  int mtile = blockIdx.x * 8 + (threadIdx.x >> 5);
  int m0    = mtile << 4;
  if (m0 >= M) return;
  int n0 = blockIdx.y << 7;

  v8f acc[8] = {v8f{}, v8f{}, v8f{}, v8f{}, v8f{}, v8f{}, v8f{}, v8f{}};
#pragma unroll
  for (int k0 = 0; k0 < HIDDIM; k0 += 32) {
    v16bf a1 = frag_bf16(A1, m0, k0, lane);
    v16bf a2 = frag_bf16(A2, m0, k0, lane);
#pragma unroll
    for (int t = 0; t < 8; ++t) {
      v16bf b1 = frag_bf16(W1, n0 + t * 16, k0, lane);
      acc[t] = __builtin_amdgcn_wmma_f32_16x16x32_bf16(
          false, a1, false, b1, (short)0, acc[t], false, false);
      v16bf b2 = frag_bf16(W2, n0 + t * 16, k0, lane);
      acc[t] = __builtin_amdgcn_wmma_f32_16x16x32_bf16(
          false, a2, false, b2, (short)0, acc[t], false, false);
    }
  }
#pragma unroll
  for (int t = 0; t < 8; ++t) {
    int n = n0 + t * 16 + (lane & 15);
#pragma unroll
    for (int v = 0; v < 8; ++v) {
      int m = m0 + ((lane >> 4) << 3) + v;
      float x = acc[t][v];
      C[(size_t)m * HIDDIM + n] = x > 0.f ? x : 0.f;
    }
  }
}

// ---------------------------------------------------------------------------
// SpMM scatter: one wave32 per edge; lane handles 8 consecutive elements.
// Gathers bf16 rows (halves the dominant memory stream), accumulates fp32
// via global_atomic_add_f32.
// ---------------------------------------------------------------------------
__global__ void spmm_scatter(const int* __restrict__ rows, const int* __restrict__ cols,
                             const float* __restrict__ vals,
                             const __bf16* __restrict__ Hsrc, float* __restrict__ Hdst,
                             int nnz) {
  int e = blockIdx.x * 8 + (threadIdx.x >> 5);
  if (e >= nnz) return;
  int lane = threadIdx.x & 31;

  // prefetch the gather row of a later edge into cache
  int e2 = e + 512;
  if (e2 < nnz) {
    int c2 = cols[e2];
    __builtin_prefetch(Hsrc + (size_t)c2 * HIDDIM + lane * 8, 0, 1);
  }

  int   r = rows[e];
  int   c = cols[e];
  float v = vals[e];
  const __bf16* src = Hsrc + (size_t)c * HIDDIM + lane * 8;   // 16B / lane
  float*        dst = Hdst + (size_t)r * HIDDIM + lane * 8;
#pragma unroll
  for (int j = 0; j < 8; ++j) {
    unsafeAtomicAdd(&dst[j], v * (float)src[j]);
  }
}

// ---------------------------------------------------------------------------
// logits[M,1] = H2 @ W_out^T + b_out ; one wave per row, fp32 reduction
// ---------------------------------------------------------------------------
__global__ void gemv_out(const float* __restrict__ H2, const float* __restrict__ Wout,
                         const float* __restrict__ bout, float* __restrict__ out, int M) {
  int row = blockIdx.x * 8 + (threadIdx.x >> 5);
  if (row >= M) return;
  int lane = threadIdx.x & 31;
  const float* p = H2 + (size_t)row * HIDDIM + lane * 8;
  const float* w = Wout + lane * 8;
  float s = 0.f;
#pragma unroll
  for (int j = 0; j < 8; ++j) s += p[j] * w[j];
#pragma unroll
  for (int off = 16; off > 0; off >>= 1) s += __shfl_xor(s, off, 32);
  if (lane == 0) out[row] = s + bout[0];
}

// ---------------------------------------------------------------------------
extern "C" void kernel_launch(void* const* d_in, const int* in_sizes, int n_in,
                              void* d_out, int out_size, void* d_ws, size_t ws_size,
                              hipStream_t stream) {
  const float* X     = (const float*)d_in[0];
  const int*   rows  = (const int*)d_in[1];
  const int*   cols  = (const int*)d_in[2];
  const float* vals  = (const float*)d_in[3];
  const float* W_in  = (const float*)d_in[4];
  const float* b_in  = (const float*)d_in[5];
  const float* W_mp1 = (const float*)d_in[6];
  const float* W_mp2 = (const float*)d_in[7];
  const float* W_out = (const float*)d_in[8];
  const float* b_out = (const float*)d_in[9];

  int M   = in_sizes[0] / HIDDIM;     // 100000 (multiple of 16)
  int nnz = in_sizes[1];              // 3.2M
  long long nelem = (long long)M * HIDDIM;

  char*  ws     = (char*)d_ws;
  size_t f32B   = (size_t)M * HIDDIM * sizeof(float);
  size_t bf16B  = (size_t)M * HIDDIM * sizeof(__bf16);
  // fp32 accumulators
  float*  AH   = (float*)(ws);                     // later reused as H2 (fp32)
  float*  A2H  = (float*)(ws + f32B);
  // bf16 ping-pong buffers
  __bf16* bf0  = (__bf16*)(ws + 2 * f32B);         // Xbf, later A2Hbf
  __bf16* bf1  = (__bf16*)(ws + 2 * f32B + bf16B); // Hbf, later AHbf
  // bf16 weights
  __bf16* Wb0  = (__bf16*)(ws + 2 * f32B + 2 * bf16B);
  __bf16* Wb1  = Wb0 + HIDDIM * HIDDIM;
  __bf16* Wb2  = Wb1 + HIDDIM * HIDDIM;
  float*  out  = (float*)d_out;
  float*  H2   = AH;                               // reuse after AH is consumed

  // scatter targets must start at zero every launch (kernels accumulate)
  hipMemsetAsync(AH,  0, f32B, stream);
  hipMemsetAsync(A2H, 0, f32B, stream);

  int cblocks = (int)((nelem / 8 + 255) / 256);
  cvt_weights<<<(HIDDIM * HIDDIM) / 256, 256, 0, stream>>>(W_in, W_mp1, W_mp2,
                                                           Wb0, Wb1, Wb2);
  cvt_f32_bf16<<<cblocks, 256, 0, stream>>>(X, bf0, nelem);          // Xbf

  int  mtiles  = (M + 15) >> 4;
  dim3 ggrid((mtiles + 7) >> 3, 2);
  gemm1_relu_bias<<<ggrid, 256, 0, stream>>>(bf0, Wb0, b_in, bf1, M); // Hbf

  int sblocks = (nnz + 7) >> 3;
  spmm_scatter<<<sblocks, 256, 0, stream>>>(rows, cols, vals, bf1, AH, nnz);
  cvt_f32_bf16<<<cblocks, 256, 0, stream>>>(AH, bf1, nelem);         // AHbf
  spmm_scatter<<<sblocks, 256, 0, stream>>>(rows, cols, vals, bf1, A2H, nnz);
  cvt_f32_bf16<<<cblocks, 256, 0, stream>>>(A2H, bf0, nelem);        // A2Hbf

  gemm2_relu<<<ggrid, 256, 0, stream>>>(bf1, bf0, Wb1, Wb2, H2, M);

  int vblocks = (M + 7) >> 3;
  gemv_out<<<vblocks, 256, 0, stream>>>(H2, W_out, b_out, out, M);
}
